// PAM_Module_66992899883780
// MI455X (gfx1250) — matrix operand order repaired and checked by reading, not compiled, not run on hardware
//
#include <hip/hip_runtime.h>

// ---------- types for CDNA5 WMMA ----------
typedef __attribute__((ext_vector_type(16))) __bf16 v16bf;
typedef __attribute__((ext_vector_type(8)))  float  v8f;

union BfOp {
  v16bf bf;
  unsigned short s[16];
  unsigned u[8];
  uint4 q[2];
};

// pack two fp32 into bf16x2 (truncate) with a single v_perm_b32
__device__ __forceinline__ unsigned pkbf(float lo, float hi) {
  return __builtin_amdgcn_perm(__builtin_bit_cast(unsigned, hi),
                               __builtin_bit_cast(unsigned, lo), 0x07060302u);
}

#define NB 4
#define CC 256
#define DD 32
#define NN 4096
#define L2E 1.44269504088896340736f

// ------------------------------------------------------------------
// Kernel 1: Q/K projections, fp32 VALU.  pq is pre-scaled by log2(e)
// so the attention softmax can use exp2 directly (bare v_exp_f32).
// Layout: bf16 [b][n][32] -> WMMA A (Q) / B (K) operand patterns.
// ------------------------------------------------------------------
__global__ void pam_proj_qk(const float* __restrict__ q, const float* __restrict__ k,
                            const float* __restrict__ Wq, const float* __restrict__ bq,
                            const float* __restrict__ Wk, const float* __restrict__ bk,
                            unsigned* __restrict__ pq, unsigned* __restrict__ pk) {
  int gid = blockIdx.x * blockDim.x + threadIdx.x;   // B*N threads
  int b = gid >> 12;
  int n = gid & (NN - 1);
  const float* qp = q + (size_t)b * (CC * NN) + n;
  const float* kp = k + (size_t)b * (CC * NN) + n;
  float aq[DD], ak[DD];
#pragma unroll
  for (int d = 0; d < DD; ++d) { aq[d] = bq[d]; ak[d] = bk[d]; }
#pragma unroll 4
  for (int c = 0; c < CC; ++c) {
    float qv = qp[(unsigned)(c * NN)];   // coalesced across lanes
    float kv = kp[(unsigned)(c * NN)];
#pragma unroll
    for (int d = 0; d < DD; ++d) {
      aq[d] = fmaf(Wq[d * CC + c], qv, aq[d]);   // uniform -> scalar loads
      ak[d] = fmaf(Wk[d * CC + c], kv, ak[d]);
    }
  }
  unsigned* oq = pq + (size_t)gid * (DD / 2);
  unsigned* ok = pk + (size_t)gid * (DD / 2);
#pragma unroll
  for (int d = 0; d < DD / 2; ++d) {
    oq[d] = pkbf(aq[2 * d] * L2E, aq[2 * d + 1] * L2E);
    ok[d] = pkbf(ak[2 * d], ak[2 * d + 1]);
  }
}

// ------------------------------------------------------------------
// Kernel 2: V projection, bf16 WMMA GEMM: pv[b] = Wv(256x256) @ v[b](256x4096)
// One wave per 16x16 tile, 8 x K=32.  Output bf16 [b][c][n].
// ------------------------------------------------------------------
__global__ void pam_proj_v(const float* __restrict__ v, const float* __restrict__ Wv,
                           const float* __restrict__ bv, unsigned short* __restrict__ pv) {
  int tid  = blockIdx.x * blockDim.x + threadIdx.x;
  int wave = tid >> 5;                 // B * 16 * 256 waves
  int lane = tid & 31;
  int b  = wave >> 12;
  int r  = wave & 4095;
  int mb = (r >> 8) << 4;              // output-channel tile base
  int nb = (r & 255) << 4;             // pixel tile base
  int lm = lane & 15;
  int hi = lane >> 4;

  const float* vb          = v  + (size_t)b * (CC * NN);   // uniform batch base
  unsigned short* pvb      = pv + (size_t)b * (CC * NN);

  v8f acc = {};
  const float* wrow = Wv + (mb + lm) * CC;   // A row (output channel)
  unsigned ncol = nb + lm;                   // B column (pixel)

#pragma unroll 1
  for (int kb = 0; kb < 8; ++kb) {
    // A operand: lanes 0-15 K=0..15, lanes 16-31 K=16..31
    int k0 = kb * 32 + hi * 16;
    BfOp A;
#pragma unroll
    for (int t = 0; t < 4; ++t) A.u[t]     = pkbf(wrow[k0 + 2 * t],      wrow[k0 + 2 * t + 1]);
#pragma unroll
    for (int t = 0; t < 4; ++t) A.u[4 + t] = pkbf(wrow[k0 + 8 + 2 * t], wrow[k0 + 9 + 2 * t]);
    // B operand: lanes 0-15 K=0..15, lanes 16-31 K=16..31, N=lane%16
    BfOp Bm;
    unsigned kbase = kb * 32 + hi * 16;
#pragma unroll
    for (int t = 0; t < 8; ++t)
      Bm.u[t] = pkbf(vb[(kbase + 2 * t) * NN + ncol],
                     vb[(kbase + 2 * t + 1) * NN + ncol]);
    acc = __builtin_amdgcn_wmma_f32_16x16x32_bf16(false, A.bf, false, Bm.bf,
                                                  (short)0, acc, false, false);
  }
#pragma unroll
  for (int j = 0; j < 8; ++j) {
    unsigned c = mb + j + hi * 8;
    float val = acc[j] + bv[c];
    unsigned ub = pkbf(val, val);
    pvb[c * NN + ncol] = (unsigned short)(ub & 0xFFFFu);
  }
}

// ------------------------------------------------------------------
// Kernel 3: flash attention.  One wave = 16 queries x 128 channels.
// 64-key rounds: 4 S-WMMAs (stride-2 key pairing -> packed b32 DS
// stores of adjacent-key probs), exp2-domain online max with per-lane
// deferred row sums, P via LDS transpose to A-operand, 16 PV WMMAs.
// Next-round K/V prefetched via global_prefetch_b8.
// ------------------------------------------------------------------
__global__ void pam_attn(const float* __restrict__ vin, const float* __restrict__ gamma,
                         const unsigned short* __restrict__ pq,
                         const unsigned short* __restrict__ pk,
                         const unsigned short* __restrict__ pv,
                         float* __restrict__ out) {
  __shared__ __align__(16) unsigned plds[8 * 16 * 32];   // per wave: 16 rows x 64 bf16 keys
  int tid  = blockIdx.x * blockDim.x + threadIdx.x;
  int wave = tid >> 5;                 // 0..2047
  int wloc = threadIdx.x >> 5;
  int lane = tid & 31;
  int b  = wave >> 9;                  // / 512
  int r  = wave & 511;
  int qb = (r >> 1) << 4;              // query tile base
  int cb = (r & 1) << 7;               // channel half: 0 or 128
  int lm = lane & 15;
  int hi = lane >> 4;

  // uniform per-batch base pointers -> 32-bit VGPR offsets (GVS addressing)
  const unsigned short* qbp = pq  + (size_t)b * (NN * DD);
  const unsigned short* kbp = pk  + (size_t)b * (NN * DD);
  const unsigned short* vbp = pv  + (size_t)b * (CC * NN);
  const float*          ibp = vin + (size_t)b * (CC * NN);
  float*                obp = out + (size_t)b * (CC * NN);

  // Persistent Q A-operand (row = qb + lane%16; contiguous K-halves, K = hi*16 + s)
  BfOp Qa;
  {
    const uint4* qp = (const uint4*)(qbp + (unsigned)((qb + lm) * DD));
    Qa.q[0] = qp[2 * hi];        // K = hi*16 .. +7
    Qa.q[1] = qp[2 * hi + 1];    // K = hi*16 + 8 .. +15
  }

  v8f acc[8];
#pragma unroll
  for (int g = 0; g < 8; ++g) acc[g] = (v8f){};
  float mrun[8], lsum[8];
#pragma unroll
  for (int j = 0; j < 8; ++j) { mrun[j] = -3.0e38f; lsum[j] = 0.f; }

  unsigned* myl = plds + wloc * 512;   // 16 rows x 32 uints (row = 64 keys)

#pragma unroll 1
  for (int mt = 0; mt < NN / 64; ++mt) {
    unsigned mb0 = mt * 64;
    // S B-operands, stride-2 key pairing: tile t covers keys
    // mb0 + (t>>1)*32 + 2*col + (t&1);  per-lane 32B contiguous loads.
    unsigned k00 = (mb0 + 2 * lm) * DD + hi * 16;
    BfOp K0, K1, K2, K3;
    {
      const uint4* p0 = (const uint4*)(kbp + k00);
      const uint4* p1 = (const uint4*)(kbp + k00 + DD);          // key +1
      const uint4* p2 = (const uint4*)(kbp + k00 + 32 * DD);     // key +32
      const uint4* p3 = (const uint4*)(kbp + k00 + 33 * DD);
      K0.q[0] = p0[0]; K0.q[1] = p0[1];
      K1.q[0] = p1[0]; K1.q[1] = p1[1];
      K2.q[0] = p2[0]; K2.q[1] = p2[1];
      K3.q[0] = p3[0]; K3.q[1] = p3[1];
    }
    // prefetch next round's K rows (speculative; dropped past the end)
    __builtin_prefetch(kbp + k00 + 64 * DD, 0, 1);
    __builtin_prefetch(kbp + k00 + 96 * DD, 0, 1);

    v8f z = {};
    v8f S0 = __builtin_amdgcn_wmma_f32_16x16x32_bf16(false, Qa.bf, false, K0.bf, (short)0, z, false, false);
    v8f S1 = __builtin_amdgcn_wmma_f32_16x16x32_bf16(false, Qa.bf, false, K1.bf, (short)0, z, false, false);
    v8f S2 = __builtin_amdgcn_wmma_f32_16x16x32_bf16(false, Qa.bf, false, K2.bf, (short)0, z, false, false);
    v8f S3 = __builtin_amdgcn_wmma_f32_16x16x32_bf16(false, Qa.bf, false, K3.bf, (short)0, z, false, false);

    // Online softmax in the exp2 domain (pq pre-scaled by log2e).
    // C layout: VGPR j <-> row (j + hi*8); row values live across the
    // 16 lanes of this half -> xor 1/2/4/8 reductions stay in-half.
#pragma unroll
    for (int j = 0; j < 8; ++j) {
      float tm = fmaxf(fmaxf(S0[j], S1[j]), fmaxf(S2[j], S3[j]));
      tm = fmaxf(tm, __shfl_xor(tm, 1, 32));
      tm = fmaxf(tm, __shfl_xor(tm, 2, 32));
      tm = fmaxf(tm, __shfl_xor(tm, 4, 32));
      tm = fmaxf(tm, __shfl_xor(tm, 8, 32));
      float nm = fmaxf(mrun[j], tm);
      float sc = exp2f(mrun[j] - nm);
      mrun[j] = nm;
      float p0 = exp2f(S0[j] - nm);
      float p1 = exp2f(S1[j] - nm);
      float p2 = exp2f(S2[j] - nm);
      float p3 = exp2f(S3[j] - nm);
      lsum[j] = lsum[j] * sc + ((p0 + p1) + (p2 + p3));  // per-lane partial rowsum
#pragma unroll
      for (int g = 0; g < 8; ++g) acc[g][j] *= sc;
      int row = j + hi * 8;
      myl[row * 32 + lm]      = pkbf(p0, p1);   // keys 2*lm, 2*lm+1
      myl[row * 32 + 16 + lm] = pkbf(p2, p3);   // keys 32+2*lm, 33+2*lm
    }

    // P A-operands from LDS (same-wave DS ops are in-order)
    BfOp Pa0, Pa1;
    {
      const unsigned* pr = myl + lm * 32;
      Pa0.q[0] = *(const uint4*)(pr + hi * 8);         // K = hi*16 .. +7
      Pa0.q[1] = *(const uint4*)(pr + 4 + hi * 8);     // K = hi*16 + 8 ..
      Pa1.q[0] = *(const uint4*)(pr + 16 + hi * 8);
      Pa1.q[1] = *(const uint4*)(pr + 20 + hi * 8);
    }

    // acc += P @ V^T, 8 channel groups x 2 key-chunks
#pragma unroll
    for (int g = 0; g < 8; ++g) {
      unsigned voff = (cb + g * 16 + lm) * NN + mb0 + hi * 16;
      const uint4* vp = (const uint4*)(vbp + voff);
      __builtin_prefetch(vbp + voff + 64, 0, 1);   // next round's V segment
      BfOp Vb0, Vb1;
      Vb0.q[0] = vp[0]; Vb0.q[1] = vp[1];
      Vb1.q[0] = vp[4]; Vb1.q[1] = vp[5];   // +32 keys = +4 uint4
      acc[g] = __builtin_amdgcn_wmma_f32_16x16x32_bf16(false, Pa0.bf, false, Vb0.bf,
                                                       (short)0, acc[g], false, false);
      acc[g] = __builtin_amdgcn_wmma_f32_16x16x32_bf16(false, Pa1.bf, false, Vb1.bf,
                                                       (short)0, acc[g], false, false);
    }
  }

  // Final rowsum reduction (once), then out = gamma * acc/rowsum + v
  float rl[8];
#pragma unroll
  for (int j = 0; j < 8; ++j) {
    float ls = lsum[j];
    ls += __shfl_xor(ls, 1, 32);
    ls += __shfl_xor(ls, 2, 32);
    ls += __shfl_xor(ls, 4, 32);
    ls += __shfl_xor(ls, 8, 32);
    rl[j] = 1.0f / ls;
  }
  float gm = gamma[0];
#pragma unroll
  for (int g = 0; g < 8; ++g) {
#pragma unroll
    for (int j = 0; j < 8; ++j) {
      unsigned c = cb + g * 16 + lm;
      unsigned n = qb + j + hi * 8;
      unsigned idx = c * NN + n;           // 32-bit offset from uniform base
      obp[idx] = gm * acc[g][j] * rl[j] + ibp[idx];
    }
  }
}

// ------------------------------------------------------------------
extern "C" void kernel_launch(void* const* d_in, const int* in_sizes, int n_in,
                              void* d_out, int out_size, void* d_ws, size_t ws_size,
                              hipStream_t stream) {
  const float* v    = (const float*)d_in[0];
  const float* k    = (const float*)d_in[1];
  const float* q    = (const float*)d_in[2];
  const float* Wq   = (const float*)d_in[3];
  const float* bq   = (const float*)d_in[4];
  const float* Wk   = (const float*)d_in[5];
  const float* bk   = (const float*)d_in[6];
  const float* Wv   = (const float*)d_in[7];
  const float* bv   = (const float*)d_in[8];
  const float* gam  = (const float*)d_in[9];
  float* out = (float*)d_out;

  // workspace: pq (1MB) | pk (1MB) | pv (8MB), all bf16
  unsigned short* pq = (unsigned short*)d_ws;
  unsigned short* pk = pq + (size_t)NB * NN * DD;
  unsigned short* pv = pk + (size_t)NB * NN * DD;

  pam_proj_qk<<<(NB * NN) / 256, 256, 0, stream>>>(q, k, Wq, bq, Wk, bk,
                                                   (unsigned*)pq, (unsigned*)pk);
  pam_proj_v<<<(NB * (CC / 16) * (NN / 16)) / 8, 256, 0, stream>>>(v, Wv, bv, pv);
  pam_attn<<<(NB * (NN / 16) * 2) / 8, 256, 0, stream>>>(v, gam, pq, pk, pv, out);
}
